// VN_Block_19164144075361
// MI455X (gfx1250) — matrix operand order repaired and checked by reading, not compiled, not run on hardware
//
#include <hip/hip_runtime.h>

// ---------------- problem constants ----------------
#define BB 4
#define NN 2048
#define CC 128
#define HH 8
#define KK 8
#define NCOLS (3 * NN)   // per-channel row length (3,N) flattened = 6144
#define EPSV 1e-6f
#define NEPS 1e-5f
#define SLOPEV 0.2f

typedef __attribute__((ext_vector_type(16))) __bf16 v16bf;
typedef __attribute__((ext_vector_type(8)))  __bf16 bf16x8;
typedef __attribute__((ext_vector_type(8)))  float  v8f;

union Frag { v16bf v; __bf16 e[16]; };

// A-fragment column permutation: makes the 16 elements a (row, half-wave)
// lane needs contiguous.  Maps 8-blocks: 0->0, 8->16, 16->8, 24->24.
__device__ __forceinline__ int permA(int kk) {
    return (kk & 7) | ((kk & 16) >> 1) | ((kk & 8) << 1);
}

// =====================================================================
// LayerNorm kernel.  mode 0: src0 = x [B,N,3C] (transpose-in), no residual.
//                    mode 1: src0,src1 = [B,C,3,N] buffers, residual add.
// =====================================================================
__global__ __launch_bounds__(128)
void ln_kernel(const float* __restrict__ src0, const float* __restrict__ src1,
               const float* __restrict__ gamma, const float* __restrict__ beta,
               float* __restrict__ outv, float* __restrict__ outn, int mode)
{
    const int c = threadIdx.x;          // channel 0..127
    const int n = blockIdx.x;           // point
    const int b = blockIdx.y;           // batch
    const size_t off = ((size_t)(b * CC + c) * 3) * NN + n;

    float vx, vy, vz;
    if (mode == 0) {
        const float* p = src0 + ((size_t)b * NN + n) * (3 * CC) + c * 3;
        vx = p[0]; vy = p[1]; vz = p[2];
    } else {
        vx = src0[off]          + src1[off];
        vy = src0[off + NN]     + src1[off + NN];
        vz = src0[off + 2 * NN] + src1[off + 2 * NN];
    }
    float nn = sqrtf(vx * vx + vy * vy + vz * vz);

    __shared__ float r1[128], r2[128];
    r1[c] = nn; r2[c] = nn * nn;
    __syncthreads();
    for (int s = 64; s > 0; s >>= 1) {
        if (c < s) { r1[c] += r1[c + s]; r2[c] += r2[c + s]; }
        __syncthreads();
    }
    float mean = r1[0] * (1.0f / CC);
    float var  = r2[0] * (1.0f / CC) - mean * mean;
    float ln   = (nn - mean) * rsqrtf(var + NEPS) * gamma[c] + beta[c];
    float fac  = ln / (nn + EPSV);

    if (outv) { outv[off] = vx; outv[off + NN] = vy; outv[off + 2 * NN] = vz; }
    outn[off]          = vx * fac;
    outn[off + NN]     = vy * fac;
    outn[off + 2 * NN] = vz * fac;
}

// =====================================================================
// Generic GEMM:  Y[b] (+)= W(M x Kc, row stride ldw) * X[b](Kc x NCOLS)
// bf16 WMMA 16x16x32, f32 accumulate.  Block = 128 thr (4 waves),
// block tile = 64(M) x 32(cols); each wave owns a 16x32 output slab.
// LDS tiles are stored in WMMA-fragment order: every lane reads its
// entire 16-element fragment as one contiguous 32B chunk (2x ds_load_b128).
// =====================================================================
__global__ __launch_bounds__(128)
void gemm_cn(const float* __restrict__ W, int ldw, int M, int Kc,
             const float* __restrict__ X, float* __restrict__ Y, int acc)
{
    __shared__ __attribute__((aligned(32))) __bf16 aF[64 * 32];  // [m][permA(k)]
    __shared__ __attribute__((aligned(32))) __bf16 bF[32 * 32];  // [j][permA(k)]

    const int tid  = threadIdx.x;
    const int lane = tid & 31, wave = tid >> 5;
    const int hi   = lane >> 4, lo = lane & 15;
    const int jt   = blockIdx.x;
    const int b    = (jt * 32) / NCOLS;
    const int j0   = (jt * 32) % NCOLS;
    const int m0   = blockIdx.y * 64;
    const float* Xb = X + (size_t)b * Kc * NCOLS;
    float*       Yb = Y + (size_t)b * M  * NCOLS;

    v8f c0 = {}; v8f c1 = {};

    for (int k0 = 0; k0 < Kc; k0 += 32) {
        __syncthreads();
        // ---- stage W tile: 256 tasks of (row, 8 consecutive k) ----
#pragma unroll
        for (int t = 0; t < 2; ++t) {
            int task = tid + 128 * t;
            int r = task >> 2, kg = (task & 3) * 8;
            const float4* src = (const float4*)(W + (size_t)(m0 + r) * ldw + k0 + kg);
            float4 w0 = src[0], w1 = src[1];
            bf16x8 v;
            v[0] = (__bf16)w0.x; v[1] = (__bf16)w0.y; v[2] = (__bf16)w0.z; v[3] = (__bf16)w0.w;
            v[4] = (__bf16)w1.x; v[5] = (__bf16)w1.y; v[6] = (__bf16)w1.z; v[7] = (__bf16)w1.w;
            *(bf16x8*)(aF + r * 32 + permA(kg)) = v;
        }
        // ---- stage X tile transposed: (col j, 8 consecutive k) ----
        {
            int jj = tid & 31, kg = (tid >> 5) * 8;
            const float* src = Xb + (size_t)(k0 + kg) * NCOLS + j0 + jj;
            bf16x8 v;
#pragma unroll
            for (int q = 0; q < 8; ++q) v[q] = (__bf16)src[(size_t)q * NCOLS];
            *(bf16x8*)(bF + jj * 32 + permA(kg)) = v;
        }
        if (k0 + 32 < Kc)   // gfx1250 global_prefetch_b8 of next X tile
            __builtin_prefetch(&Xb[(size_t)(k0 + 32) * NCOLS + j0], 0, 1);
        __syncthreads();

        Frag af, b0, b1;
        af.v = *(const v16bf*)(aF + (wave * 16 + lo) * 32 + hi * 16);
        b0.v = *(const v16bf*)(bF + lo * 32 + hi * 16);
        b1.v = *(const v16bf*)(bF + (16 + lo) * 32 + hi * 16);
        c0 = __builtin_amdgcn_wmma_f32_16x16x32_bf16(false, af.v, false, b0.v,
                                                     (short)0, c0, false, false);
        c1 = __builtin_amdgcn_wmma_f32_16x16x32_bf16(false, af.v, false, b1.v,
                                                     (short)0, c1, false, false);
    }

#pragma unroll
    for (int r = 0; r < 8; ++r) {                       // C layout: row r+hi*8
        int mrow = m0 + wave * 16 + r + hi * 8;
        size_t off = (size_t)mrow * NCOLS + j0 + lo;
        if (acc) { Yb[off] += c0[r]; Yb[off + 16] += c1[r]; }
        else     { Yb[off]  = c0[r]; Yb[off + 16]  = c1[r]; }
    }
}

// =====================================================================
// Flash-style VN attention. Q/K/V in [B,C,3,N]; head h occupies the
// contiguous 48 rows [ (b*C+h*16)*3 , +48 ) of length N.  4 waves/block,
// each wave = 16 queries, loops key tiles of 32 with online softmax.
// LDS tiles stored fragment-contiguous (see gemm_cn).
// =====================================================================
__global__ __launch_bounds__(128)
void attn_kernel(const float* __restrict__ Q, const float* __restrict__ Kf,
                 const float* __restrict__ Vf, float* __restrict__ O)
{
    __shared__ __attribute__((aligned(32))) __bf16 kF[32 * 64];      // [key j][feat kk]
    __shared__ __attribute__((aligned(32))) __bf16 vF[64 * 32];      // [feat f][permA(key)]
    __shared__ __attribute__((aligned(32))) __bf16 pF[4][16 * 32];   // [q row][permA(key)]

    const int tid  = threadIdx.x, lane = tid & 31, wave = tid >> 5;
    const int hi   = lane >> 4,  lo   = lane & 15;
    const int bh   = blockIdx.y, b = bh / HH, h = bh % HH;
    const int q0   = blockIdx.x * 64 + wave * 16;
    const size_t baseC = ((size_t)(b * CC + h * 16)) * 3 * NN;

    Frag qa0, qa1;
    {
        const int row = q0 + lo;
#pragma unroll
        for (int e = 0; e < 16; ++e) {
            int k0 = e + (hi << 4);
            int k1 = k0 + 32;
            qa0.e[e] = (__bf16)Q[baseC + (size_t)k0 * NN + row];
            qa1.e[e] = (__bf16)((k1 < 48) ? Q[baseC + (size_t)k1 * NN + row] : 0.f);
        }
    }

    float m[8], l[8];
    v8f o0 = {}, o1 = {}, o2 = {};
#pragma unroll
    for (int r = 0; r < 8; ++r) { m[r] = -3.0e38f; l[r] = 0.f; }

    for (int kt = 0; kt < NN / 32; ++kt) {
        const int j0 = kt * 32;
        __syncthreads();
        // ---- stage K tile transposed: (key j, 8 consecutive feats) ----
#pragma unroll
        for (int t = 0; t < 2; ++t) {
            int task = tid + 128 * t;
            int j = task & 31, kg = (task >> 5) * 8;     // kg in {0..56}
            bf16x8 v;
#pragma unroll
            for (int q = 0; q < 8; ++q) {
                int kk = kg + q;
                v[q] = (__bf16)((kk < 48) ? Kf[baseC + (size_t)kk * NN + j0 + j] : 0.f);
            }
            *(bf16x8*)(kF + j * 64 + kg) = v;
        }
        // ---- stage V tile: (feat f, 8 consecutive keys) ----
#pragma unroll
        for (int t = 0; t < 2; ++t) {
            int task = tid + 128 * t;
            int f = task >> 2, kg = (task & 3) * 8;
            bf16x8 v;
            if (f < 48) {
                const float* src = Vf + baseC + (size_t)f * NN + j0 + kg;
                const float4* s4 = (const float4*)src;
                float4 a = s4[0], c = s4[1];
                v[0] = (__bf16)a.x; v[1] = (__bf16)a.y; v[2] = (__bf16)a.z; v[3] = (__bf16)a.w;
                v[4] = (__bf16)c.x; v[5] = (__bf16)c.y; v[6] = (__bf16)c.z; v[7] = (__bf16)c.w;
            } else {
#pragma unroll
                for (int q = 0; q < 8; ++q) v[q] = (__bf16)0.f;
            }
            *(bf16x8*)(vF + f * 32 + permA(kg)) = v;
        }
        __syncthreads();

        v8f s0 = {}, s1 = {};
        {
            Frag f0, f1;
            f0.v = *(const v16bf*)(kF + lo * 64 + hi * 16);
            f1.v = *(const v16bf*)(kF + lo * 64 + 32 + hi * 16);
            s0 = __builtin_amdgcn_wmma_f32_16x16x32_bf16(false, qa0.v, false, f0.v,
                                                         (short)0, s0, false, false);
            s0 = __builtin_amdgcn_wmma_f32_16x16x32_bf16(false, qa1.v, false, f1.v,
                                                         (short)0, s0, false, false);
            f0.v = *(const v16bf*)(kF + (16 + lo) * 64 + hi * 16);
            f1.v = *(const v16bf*)(kF + (16 + lo) * 64 + 32 + hi * 16);
            s1 = __builtin_amdgcn_wmma_f32_16x16x32_bf16(false, qa0.v, false, f0.v,
                                                         (short)0, s1, false, false);
            s1 = __builtin_amdgcn_wmma_f32_16x16x32_bf16(false, qa1.v, false, f1.v,
                                                         (short)0, s1, false, false);
        }

        // online softmax per row (row r lives in slot r of a 16-lane half)
#pragma unroll
        for (int r = 0; r < 8; ++r) {
            float a0 = s0[r] * 0.25f, a1 = s1[r] * 0.25f;   // scale = 16^-0.5
            float mx = fmaxf(a0, a1);
#pragma unroll
            for (int d = 1; d < 16; d <<= 1) mx = fmaxf(mx, __shfl_xor(mx, d, 32));
            float mnew  = fmaxf(m[r], mx);
            float alpha = __expf(m[r] - mnew);
            m[r] = mnew;
            float p0 = __expf(a0 - mnew), p1 = __expf(a1 - mnew);
            float ps = p0 + p1;
#pragma unroll
            for (int d = 1; d < 16; d <<= 1) ps += __shfl_xor(ps, d, 32);
            l[r] = l[r] * alpha + ps;
            o0[r] *= alpha; o1[r] *= alpha; o2[r] *= alpha;
            int rl = r + hi * 8;
            pF[wave][rl * 32 + permA(lo)]      = (__bf16)p0;
            pF[wave][rl * 32 + permA(16 + lo)] = (__bf16)p1;
        }
        __syncthreads();   // uniform; makes own-wave pF writes visible

        {
            Frag pa, vb;
            pa.v = *(const v16bf*)(pF[wave] + lo * 32 + hi * 16);
            vb.v = *(const v16bf*)(vF + lo * 32 + hi * 16);
            o0 = __builtin_amdgcn_wmma_f32_16x16x32_bf16(false, pa.v, false, vb.v,
                                                         (short)0, o0, false, false);
            vb.v = *(const v16bf*)(vF + (16 + lo) * 32 + hi * 16);
            o1 = __builtin_amdgcn_wmma_f32_16x16x32_bf16(false, pa.v, false, vb.v,
                                                         (short)0, o1, false, false);
            vb.v = *(const v16bf*)(vF + (32 + lo) * 32 + hi * 16);
            o2 = __builtin_amdgcn_wmma_f32_16x16x32_bf16(false, pa.v, false, vb.v,
                                                         (short)0, o2, false, false);
        }
    }

#pragma unroll
    for (int r = 0; r < 8; ++r) {
        int qrow = q0 + r + hi * 8;
        float inv = 1.0f / l[r];
        O[baseC + (size_t)(lo)      * NN + qrow] = o0[r] * inv;
        O[baseC + (size_t)(16 + lo) * NN + qrow] = o1[r] * inv;
        O[baseC + (size_t)(32 + lo) * NN + qrow] = o2[r] * inv;
    }
}

// =====================================================================
// Derived weight diffs:  wd[o,c] = W[o, 128+c] - W[o, c]   (ldw = 256)
// =====================================================================
__global__ void wdiff(const float* __restrict__ W1f, const float* __restrict__ W1d,
                      float* __restrict__ wd1f, float* __restrict__ wd1d)
{
    int i = blockIdx.x * 256 + threadIdx.x;
    if (i >= CC * CC) return;
    int o = i >> 7, c = i & 127;
    wd1f[i] = W1f[o * 256 + 128 + c] - W1f[o * 256 + c];
    wd1d[i] = W1d[o * 256 + 128 + c] - W1d[o * 256 + c];
}

// =====================================================================
// VNBatchNorm stats, graph-feature branch: p = g[.., knn(j)] + u[.., n]
// stats[c] += sum(norm), stats[256+c] += sum(norm^2)
// =====================================================================
__global__ __launch_bounds__(256)
void bn_stats_gather(const float* __restrict__ g, const float* __restrict__ u,
                     const int* __restrict__ knn, float* __restrict__ stats)
{
    const int c = blockIdx.x, b = blockIdx.y;
    const size_t base = (size_t)(b * CC + c) * 3 * NN;
    float s1 = 0.f, s2 = 0.f;
    for (int i = threadIdx.x; i < NN * KK; i += 256) {
        int n = i & (NN - 1), k = i / NN;
        int j = knn[((size_t)b * KK + k) * NN + n] - b * NN;
        float px = g[base + j]          + u[base + n];
        float py = g[base + NN + j]     + u[base + NN + n];
        float pz = g[base + 2 * NN + j] + u[base + 2 * NN + n];
        float nn = sqrtf(px * px + py * py + pz * pz) + EPSV;
        s1 += nn; s2 += nn * nn;
    }
    __shared__ float r1[256], r2[256];
    r1[threadIdx.x] = s1; r2[threadIdx.x] = s2;
    __syncthreads();
    for (int s = 128; s > 0; s >>= 1) {
        if (threadIdx.x < s) { r1[threadIdx.x] += r1[threadIdx.x + s]; r2[threadIdx.x] += r2[threadIdx.x + s]; }
        __syncthreads();
    }
    if (threadIdx.x == 0) { atomicAdd(&stats[c], r1[0]); atomicAdd(&stats[256 + c], r2[0]); }
}

// =====================================================================
// BN + direction-gated leaky + mean over K neighbors -> knn_f [B,C,3,N]
// =====================================================================
__global__ __launch_bounds__(256)
void bn_leaky_gather(const float* __restrict__ g,  const float* __restrict__ u,
                     const float* __restrict__ gd, const float* __restrict__ ud,
                     const int* __restrict__ knn,  const float* __restrict__ stats,
                     const float* __restrict__ gamma, const float* __restrict__ beta,
                     float* __restrict__ outb)
{
    const int n = blockIdx.x * 256 + threadIdx.x;
    const int c = blockIdx.y, b = blockIdx.z;
    const size_t base = (size_t)(b * CC + c) * 3 * NN;
    const float cnt  = (float)BB * NN * KK;
    const float mean = stats[c] / cnt;
    const float var  = stats[256 + c] / cnt - mean * mean;
    const float is   = rsqrtf(var + NEPS);
    const float ga = gamma[c], be = beta[c];
    const float ux = u[base + n],  uy = u[base + NN + n],  uz = u[base + 2 * NN + n];
    const float wx = ud[base + n], wy = ud[base + NN + n], wz = ud[base + 2 * NN + n];
    float ax = 0.f, ay = 0.f, az = 0.f;
#pragma unroll
    for (int k = 0; k < KK; ++k) {
        int j = knn[((size_t)b * KK + k) * NN + n] - b * NN;
        float px = g[base + j] + ux, py = g[base + NN + j] + uy, pz = g[base + 2 * NN + j] + uz;
        float nn = sqrtf(px * px + py * py + pz * pz) + EPSV;
        float nb = (nn - mean) * is * ga + be;
        float f  = nb / nn;
        float qx = px * f, qy = py * f, qz = pz * f;
        float vx = gd[base + j] + wx, vy = gd[base + NN + j] + wy, vz = gd[base + 2 * NN + j] + wz;
        float dot = qx * vx + qy * vy + qz * vz;
        float dsq = vx * vx + vy * vy + vz * vz;
        float t   = dot / (dsq + EPSV);
        float rx = (dot >= 0.f) ? qx : qx - t * vx;
        float ry = (dot >= 0.f) ? qy : qy - t * vy;
        float rz = (dot >= 0.f) ? qz : qz - t * vz;
        ax += SLOPEV * qx + (1.f - SLOPEV) * rx;
        ay += SLOPEV * qy + (1.f - SLOPEV) * ry;
        az += SLOPEV * qz + (1.f - SLOPEV) * rz;
    }
    outb[base + n]          = ax * (1.0f / KK);
    outb[base + NN + n]     = ay * (1.0f / KK);
    outb[base + 2 * NN + n] = az * (1.0f / KK);
}

// =====================================================================
// Plain BN stats over a [B,Cc,3,N] buffer (per-channel over b,n)
// =====================================================================
__global__ __launch_bounds__(256)
void bn_stats_plain(const float* __restrict__ p, float* __restrict__ stats, int Cc)
{
    const int c = blockIdx.x, b = blockIdx.y;
    const size_t base = (size_t)(b * Cc + c) * 3 * NN;
    float s1 = 0.f, s2 = 0.f;
    for (int n = threadIdx.x; n < NN; n += 256) {
        float px = p[base + n], py = p[base + NN + n], pz = p[base + 2 * NN + n];
        float nn = sqrtf(px * px + py * py + pz * pz) + EPSV;
        s1 += nn; s2 += nn * nn;
    }
    __shared__ float r1[256], r2[256];
    r1[threadIdx.x] = s1; r2[threadIdx.x] = s2;
    __syncthreads();
    for (int s = 128; s > 0; s >>= 1) {
        if (threadIdx.x < s) { r1[threadIdx.x] += r1[threadIdx.x + s]; r2[threadIdx.x] += r2[threadIdx.x + s]; }
        __syncthreads();
    }
    if (threadIdx.x == 0) { atomicAdd(&stats[c], r1[0]); atomicAdd(&stats[256 + c], r2[0]); }
}

// =====================================================================
// Plain BN + leaky combine of p,d buffers -> out [B,Cc,3,N]
// =====================================================================
__global__ __launch_bounds__(256)
void bn_leaky_plain(const float* __restrict__ p, const float* __restrict__ d,
                    const float* __restrict__ stats,
                    const float* __restrict__ gamma, const float* __restrict__ beta,
                    float* __restrict__ outb, int Cc)
{
    const int n = blockIdx.x * 256 + threadIdx.x;
    const int c = blockIdx.y, b = blockIdx.z;
    const size_t base = (size_t)(b * Cc + c) * 3 * NN;
    const float cnt  = (float)BB * NN;
    const float mean = stats[c] / cnt;
    const float var  = stats[256 + c] / cnt - mean * mean;
    const float is   = rsqrtf(var + NEPS);
    float px = p[base + n], py = p[base + NN + n], pz = p[base + 2 * NN + n];
    float nn = sqrtf(px * px + py * py + pz * pz) + EPSV;
    float nb = (nn - mean) * is * gamma[c] + beta[c];
    float f  = nb / nn;
    float qx = px * f, qy = py * f, qz = pz * f;
    float vx = d[base + n], vy = d[base + NN + n], vz = d[base + 2 * NN + n];
    float dot = qx * vx + qy * vy + qz * vz;
    float dsq = vx * vx + vy * vy + vz * vz;
    float t   = dot / (dsq + EPSV);
    float rx = (dot >= 0.f) ? qx : qx - t * vx;
    float ry = (dot >= 0.f) ? qy : qy - t * vy;
    float rz = (dot >= 0.f) ? qz : qz - t * vz;
    outb[base + n]          = SLOPEV * qx + (1.f - SLOPEV) * rx;
    outb[base + NN + n]     = SLOPEV * qy + (1.f - SLOPEV) * ry;
    outb[base + 2 * NN + n] = SLOPEV * qz + (1.f - SLOPEV) * rz;
}

// =====================================================================
// Final residual + transpose back to [B,N,3C]
// =====================================================================
__global__ void final_out(const float* __restrict__ a, const float* __restrict__ b2,
                          float* __restrict__ out)
{
    int idx = blockIdx.x * 256 + threadIdx.x;
    if (idx >= BB * NN * 3 * CC) return;
    int d = idx % 3;
    int c = (idx / 3) % CC;
    int n = (idx / (3 * CC)) % NN;
    int b = idx / (3 * CC * NN);
    size_t off = ((size_t)(b * CC + c) * 3 + d) * NN + n;
    out[idx] = a[off] + b2[off];
}

// =====================================================================
extern "C" void kernel_launch(void* const* d_in, const int* in_sizes, int n_in,
                              void* d_out, int out_size, void* d_ws, size_t ws_size,
                              hipStream_t stream)
{
    (void)in_sizes; (void)n_in; (void)out_size; (void)ws_size;
    const float* x   = (const float*)d_in[0];
    const float* g1  = (const float*)d_in[1];
    const float* b1  = (const float*)d_in[2];
    const float* Wq  = (const float*)d_in[3];
    const float* Wk  = (const float*)d_in[4];
    const float* Wv  = (const float*)d_in[5];
    const float* Wo  = (const float*)d_in[6];
    const float* W1f = (const float*)d_in[7];
    const float* W1d = (const float*)d_in[8];
    const float* g1b = (const float*)d_in[9];
    const float* b1b = (const float*)d_in[10];
    const float* W2  = (const float*)d_in[11];
    const float* g2  = (const float*)d_in[12];
    const float* b2  = (const float*)d_in[13];
    const float* W3f = (const float*)d_in[14];
    const float* W3d = (const float*)d_in[15];
    const float* g3b = (const float*)d_in[16];
    const float* b3b = (const float*)d_in[17];
    const float* W4f = (const float*)d_in[18];
    const float* W4d = (const float*)d_in[19];
    const float* g4b = (const float*)d_in[20];
    const float* b4b = (const float*)d_in[21];
    const int*   knn = (const int*)d_in[22];
    float* out = (float*)d_out;
    float* ws  = (float*)d_ws;

    const size_t SZ = (size_t)BB * CC * 3 * NN;   // one [B,C,3,N] buffer
    float* vnx = ws;             float* nrm = ws + SZ;
    float* qb  = ws + 2 * SZ;    float* kb  = ws + 3 * SZ;
    float* vb  = ws + 4 * SZ;    float* ao  = ws + 5 * SZ;
    float* e1  = ws + 6 * SZ;    float* f1  = ws + 7 * SZ;
    float* p1  = ws + 8 * SZ;    // 2*SZ (Cc=256)
    float* d1  = ws + 10 * SZ;   // 2*SZ
    float* hh  = ws + 12 * SZ;   // 2*SZ
    float* wd1f  = ws + 14 * SZ;
    float* wd1d  = wd1f + CC * CC;
    float* stats = wd1d + CC * CC;   // 512 floats

    const dim3 gLN(NN, BB), bLN(128);
    const dim3 gG128(BB * NCOLS / 32, CC / 64), gG256(BB * NCOLS / 32, 256 / 64);
    const dim3 bG(128);

    // 1. transpose-in + LayerNorm1
    ln_kernel<<<gLN, bLN, 0, stream>>>(x, nullptr, g1, b1, vnx, nrm, 0);
    wdiff<<<(CC * CC + 255) / 256, 256, 0, stream>>>(W1f, W1d, wd1f, wd1d);

    // 2. attention projections
    gemm_cn<<<gG128, bG, 0, stream>>>(Wq, CC, CC, CC, nrm, qb, 0);
    gemm_cn<<<gG128, bG, 0, stream>>>(Wk, CC, CC, CC, nrm, kb, 0);
    gemm_cn<<<gG128, bG, 0, stream>>>(Wv, CC, CC, CC, nrm, vb, 0);

    // 3. flash attention
    attn_kernel<<<dim3(NN / 64, BB * HH), dim3(128), 0, stream>>>(qb, kb, vb, ao);

    // 4. output projection (qb now dead -> reuse)
    gemm_cn<<<gG128, bG, 0, stream>>>(Wo, CC, CC, CC, ao, qb, 0);

    // 5. graph-feature branch factored: g = W1f_A@nrm, u = (W1f_B-W1f_A)@nrm, etc.
    gemm_cn<<<gG128, bG, 0, stream>>>(W1f,  2 * CC, CC, CC, nrm, kb, 0);   // g
    gemm_cn<<<gG128, bG, 0, stream>>>(wd1f, CC,     CC, CC, nrm, vb, 0);   // u
    gemm_cn<<<gG128, bG, 0, stream>>>(W1d,  2 * CC, CC, CC, nrm, ao, 0);   // gd
    gemm_cn<<<gG128, bG, 0, stream>>>(wd1d, CC,     CC, CC, nrm, e1, 0);   // ud

    hipMemsetAsync(stats, 0, 512 * sizeof(float), stream);
    bn_stats_gather<<<dim3(CC, BB), 256, 0, stream>>>(kb, vb, knn, stats);
    bn_leaky_gather<<<dim3(NN / 256, CC, BB), 256, 0, stream>>>(kb, vb, ao, e1, knn,
                                                                stats, g1b, b1b, f1);

    // 6. x1 = W2_A @ attn_out + W2_B @ knn_f   (into kb)
    gemm_cn<<<gG128, bG, 0, stream>>>(W2,       2 * CC, CC, CC, qb, kb, 0);
    gemm_cn<<<gG128, bG, 0, stream>>>(W2 + CC,  2 * CC, CC, CC, f1, kb, 1);

    // 7. residual + LayerNorm2:  vnx2 -> vb, nx -> ao
    ln_kernel<<<gLN, bLN, 0, stream>>>(vnx, kb, g2, b2, vb, ao, 1);

    // 8. vn_leaky(nx; W3f, W3d)  (Cout = 256)
    gemm_cn<<<gG256, bG, 0, stream>>>(W3f, CC, 2 * CC, CC, ao, p1, 0);
    gemm_cn<<<gG256, bG, 0, stream>>>(W3d, CC, 2 * CC, CC, ao, d1, 0);
    hipMemsetAsync(stats, 0, 512 * sizeof(float), stream);
    bn_stats_plain<<<dim3(2 * CC, BB), 256, 0, stream>>>(p1, stats, 2 * CC);
    bn_leaky_plain<<<dim3(NN / 256, 2 * CC, BB), 256, 0, stream>>>(p1, d1, stats,
                                                                   g3b, b3b, hh, 2 * CC);

    // 9. vn_leaky(hh; W4f, W4d)  (Cout = 128, K = 256)
    gemm_cn<<<gG128, bG, 0, stream>>>(W4f, 2 * CC, CC, 2 * CC, hh, e1, 0);
    gemm_cn<<<gG128, bG, 0, stream>>>(W4d, 2 * CC, CC, 2 * CC, hh, f1, 0);
    hipMemsetAsync(stats, 0, 512 * sizeof(float), stream);
    bn_stats_plain<<<dim3(CC, BB), 256, 0, stream>>>(e1, stats, CC);
    bn_leaky_plain<<<dim3(NN / 256, CC, BB), 256, 0, stream>>>(e1, f1, stats,
                                                               g4b, b4b, qb, CC);

    // 10. final residual + transpose out
    final_out<<<(BB * NN * 3 * CC + 255) / 256, 256, 0, stream>>>(vb, qb, out);
}
